// LSTM_70824010711234
// MI455X (gfx1250) — matrix operand order repaired and checked by reading, not compiled, not run on hardware
//
#include <hip/hip_runtime.h>

// ---------------- problem constants ----------------
constexpr int Bsz = 8, Ssz = 256, Hsz = 1024, Vsz = 32000, Lsz = 2, Rsz = 5;
constexpr float EPSF = 1e-6f;

// ---------------- WMMA vector types ----------------
typedef __attribute__((ext_vector_type(16))) __bf16 v16bf;
typedef __attribute__((ext_vector_type(8)))  float  v8f;
typedef int mog_v4i __attribute__((vector_size(16)));   // matches async-LDS builtin param

#if __has_builtin(__builtin_amdgcn_global_load_async_to_lds_b128) && \
    __has_builtin(__builtin_amdgcn_s_wait_asynccnt)
#define MOG_ASYNC_LDS 1
#else
#define MOG_ASYNC_LDS 0
#endif

// A-matrix (16x32 bf16) per-lane K offset inside a 32-wide K block (ISA 7.12.2).
// lanes 0-15: j=0..7 -> k=j ; j=8..15 -> k=j+8.  lanes 16-31: +8.
__device__ __forceinline__ int a_koff(int lane, int j) {
  int base = (j < 8) ? j : (j + 8);
  return base + ((lane & 16) ? 8 : 0);
}

// Inverse map: element (m, k) of an [8,K] A matrix -> flat index in packed
// A-fragment buffer laid out as packed[(kt*32 + lane)*16 + j].
__device__ __forceinline__ long packA_idx(int m, int k) {
  int kt = k >> 5, ko = k & 31, g = ko >> 3;           // g = which 8-wide K group
  int lane = m + ((g & 1) ? 16 : 0);                   // groups 1,3 live in lanes 16-31
  int j = (g == 0) ? ko : ((g == 3) ? ko - 16 : ko - 8);
  return ((long)kt * 32 + lane) * 16 + j;
}

__device__ __forceinline__ float sigmoidf_(float x) { return 1.f / (1.f + __expf(-x)); }

// ---------------- pack weight matrix W[K][N] (or W[N][K] if transB) into
// WMMA B-operand tiles: packed[((nt*KT + kt)*32 + lane)*16 + j] ----------------
__global__ void mog_pack_B(const float* __restrict__ W, __bf16* __restrict__ outp,
                           long total, int K, int N, int transB) {
  long stride = (long)gridDim.x * blockDim.x;
  int KT = K >> 5;
  for (long idx = (long)blockIdx.x * blockDim.x + threadIdx.x; idx < total; idx += stride) {
    int  j    = (int)(idx & 15);
    int  lane = (int)((idx >> 4) & 31);
    long rest = idx >> 9;
    int  kt   = (int)(rest % KT);
    long nt   = rest / KT;
    int  k    = (kt << 5) + a_koff(lane, j);
    int  n    = (int)(nt << 4) + (lane & 15);
    float v = transB ? W[(long)n * K + k] : W[(long)k * N + n];
    outp[idx] = (__bf16)v;
  }
}

// pack ys[M=2048][H] fp32 into WMMA A-operand tiles (KT = H/32)
__global__ void mog_pack_A(const float* __restrict__ Y, __bf16* __restrict__ outp, long total) {
  long stride = (long)gridDim.x * blockDim.x;
  const int KT = Hsz >> 5;
  for (long idx = (long)blockIdx.x * blockDim.x + threadIdx.x; idx < total; idx += stride) {
    int  j    = (int)(idx & 15);
    int  lane = (int)((idx >> 4) & 31);
    long rest = idx >> 9;
    int  kt   = (int)(rest % KT);
    long mt   = rest / KT;
    int  m    = (int)(mt << 4) + (lane & 15);
    int  k    = (kt << 5) + a_koff(lane, j);
    outp[idx] = (__bf16)Y[(long)m * Hsz + k];
  }
}

// ---------------- M=8 recurrent GEMM: OUT = act(A1@W1 [+ A2@W2] + biases) ----------------
// A operands arrive PRE-PACKED in WMMA A-fragment layout (rows 8..15 pre-zeroed).
// mode 0: out = v ;  1: out = relu(v) ;  2: out = mul * 2*sigmoid(v) ;  3: out = add1+add2+v
// outP (optional): bf16 A-fragment-layout copy of the fp32 output (for chained GEMMs).
__global__ __launch_bounds__(256) void mog_smallgemm(
    const __bf16* __restrict__ A1, const __bf16* __restrict__ W1, int K1,
    const __bf16* __restrict__ A2, const __bf16* __restrict__ W2, int K2,
    const float* __restrict__ b1, const float* __restrict__ b2, const float* __restrict__ b3,
    const float* __restrict__ add1, const float* __restrict__ add2,
    const float* __restrict__ mul, float* __restrict__ out, __bf16* __restrict__ outP,
    int N, int mode) {
  int gtid = blockIdx.x * blockDim.x + threadIdx.x;
  int wave = gtid >> 5;
  int lane = gtid & 31;
  int nt = wave;
  if (nt * 16 >= N) return;             // wave-uniform: EXEC stays all-ones for WMMA
  v8f acc = {};
  {
    int KT = K1 >> 5;
    const v16bf* aP = reinterpret_cast<const v16bf*>(A1) + lane;
    const v16bf* bP = reinterpret_cast<const v16bf*>(W1) + ((long)nt * KT * 32 + lane);
#pragma unroll 4
    for (int kt = 0; kt < KT; ++kt) {
      v16bf a = aP[(long)kt * 32];
      v16bf b = bP[(long)kt * 32];
      acc = __builtin_amdgcn_wmma_f32_16x16x32_bf16(false, a, false, b, (short)0, acc, false, false);
    }
  }
  if (A2) {
    int KT = K2 >> 5;
    const v16bf* aP = reinterpret_cast<const v16bf*>(A2) + lane;
    const v16bf* bP = reinterpret_cast<const v16bf*>(W2) + ((long)nt * KT * 32 + lane);
#pragma unroll 4
    for (int kt = 0; kt < KT; ++kt) {
      v16bf a = aP[(long)kt * 32];
      v16bf b = bP[(long)kt * 32];
      acc = __builtin_amdgcn_wmma_f32_16x16x32_bf16(false, a, false, b, (short)0, acc, false, false);
    }
  }
  if (lane >= 16) return;               // C rows 8..15 live in lanes 16..31: padding, discard
  int n = nt * 16 + lane;
  float bias = 0.f;
  if (b1) bias += b1[n];
  if (b2) bias += b2[n];
  if (b3) bias += b3[n];
#pragma unroll
  for (int r = 0; r < 8; ++r) {
    float v = acc[r] + bias;
    long o = (long)r * N + n;
    float vv;
    if (mode == 0)      vv = v;
    else if (mode == 1) vv = fmaxf(v, 0.f);
    else if (mode == 2) vv = mul[o] * (2.f * sigmoidf_(v));
    else                vv = add1[o] + add2[o] + v;
    out[o] = vv;
    if (outP) outP[packA_idx(r, n)] = (__bf16)vv;
  }
}

// ---------------- LayerNorm over H=1024 for 8 rows (one wave per row) ----------------
// Writes fp32 result AND bf16 A-fragment-layout copy.
__global__ __launch_bounds__(256) void mog_ln8(const float* __restrict__ x,
                                               const float* __restrict__ gamma,
                                               const float* __restrict__ beta,
                                               float* __restrict__ out,
                                               __bf16* __restrict__ outP) {
  int w = threadIdx.x >> 5, lane = threadIdx.x & 31;
  const float* row = x + (long)w * Hsz;
  float s = 0.f;
  for (int i = lane; i < Hsz; i += 32) s += row[i];
#pragma unroll
  for (int o = 16; o; o >>= 1) s += __shfl_xor(s, o, 32);
  float mu = s / Hsz;
  float vv = 0.f;
  for (int i = lane; i < Hsz; i += 32) { float d = row[i] - mu; vv += d * d; }
#pragma unroll
  for (int o = 16; o; o >>= 1) vv += __shfl_xor(vv, o, 32);
  float inv = 1.f / (sqrtf(vv / Hsz) + EPSF);
  for (int i = lane; i < Hsz; i += 32) {
    float r = gamma[i] * ((row[i] - mu) * inv) + beta[i];
    out[(long)w * Hsz + i] = r;
    outP[packA_idx(w, i)] = (__bf16)r;
  }
}

// ---------------- LSTM cell pointwise (also emits packed-A copy of h_next) ----------------
__global__ __launch_bounds__(256) void mog_cell(const float* __restrict__ lin,
                                                float* __restrict__ c, float* __restrict__ h,
                                                __bf16* __restrict__ hP) {
  int idx = blockIdx.x * 256 + threadIdx.x;   // 0..8191
  int b = idx >> 10, j = idx & 1023;
  const float* lr = lin + (long)b * 4 * Hsz;
  float ig = sigmoidf_(lr[j]);
  float fg = sigmoidf_(lr[Hsz + j]);
  float og = sigmoidf_(lr[2 * Hsz + j]);
  float g  = tanhf(lr[3 * Hsz + j]);
  float cn = fg * c[idx] + ig * g;
  float hn = og * tanhf(cn);
  c[idx] = cn;
  h[idx] = hn;
  hP[packA_idx(b, j)] = (__bf16)hn;
}

// ---------------- embedding gather:  Xe[t][b][h] = embed[x[b][t]][h] ----------------
__global__ void mog_gather(const int* __restrict__ ids, const float* __restrict__ embed,
                           float* __restrict__ Xe, long total) {
  long stride = (long)gridDim.x * blockDim.x;
  for (long idx = (long)blockIdx.x * blockDim.x + threadIdx.x; idx < total; idx += stride) {
    int t  = (int)(idx >> 13);           // /(8*1024)
    int r  = (int)(idx & 8191);
    int b  = r >> 10, hh = r & 1023;
    int tok = ids[(long)b * Ssz + t];
    Xe[idx] = embed[(long)tok * Hsz + hh];
  }
}

__global__ void mog_zero(float* __restrict__ p, long total) {
  long stride = (long)gridDim.x * blockDim.x;
  for (long i = (long)blockIdx.x * blockDim.x + threadIdx.x; i < total; i += stride) p[i] = 0.f;
}

// ---------------- logits: [2048,1024] x [1024,32000] bf16 WMMA ----------------
// block = 8 waves; block tile 64(M) x 128(N); wave -> 16(M) x 64(N) = 4 C fragments.
// The 8 B fragments per K-step are staged in LDS (double-buffered) so the 4 waves
// sharing each wave_n read them from LDS instead of 4x-redundant global traffic.
// Staging uses GLOBAL_LOAD_ASYNC_TO_LDS_B128 (ASYNCcnt) when the toolchain exposes
// it, else a synchronous load+ds_store fallback.
__global__ __launch_bounds__(256) void mog_logits(const __bf16* __restrict__ Ap,
                                                  const __bf16* __restrict__ Bp,
                                                  float* __restrict__ out) {
  const int KT = Hsz >> 5;  // 32
  __shared__ __bf16 btile[2][8][512];   // [buf][nt-in-block][lane*16] = 8KB per buffer
  int w = threadIdx.x >> 5, lane = threadIdx.x & 31;
  int wave_m = w & 3, wave_n = w >> 2;
  int mt  = blockIdx.y * 4 + wave_m;      // 0..127
  int ntB = blockIdx.x * 8;               // first of 8 N-tiles handled by this block
  const v16bf* aP = reinterpret_cast<const v16bf*>(Ap) + ((long)mt * KT * 32 + lane);
  // wave w stages B fragments of tile (ntB + w): 32B per lane per K-step
  const v16bf* bStage = reinterpret_cast<const v16bf*>(Bp) + ((long)(ntB + w) * KT * 32 + lane);

  auto stage = [&](int kt, int buf) {
    const char* g = (const char*)(bStage + (long)kt * 32);
    char*       l = (char*)&btile[buf][w][(long)lane * 16];
#if MOG_ASYNC_LDS
    __builtin_amdgcn_global_load_async_to_lds_b128(
        (__attribute__((address_space(1))) mog_v4i*)(size_t)g,
        (__attribute__((address_space(3))) mog_v4i*)(size_t)l, 0, 0);
    __builtin_amdgcn_global_load_async_to_lds_b128(
        (__attribute__((address_space(1))) mog_v4i*)(size_t)(g + 16),
        (__attribute__((address_space(3))) mog_v4i*)(size_t)(l + 16), 0, 0);
#else
    *reinterpret_cast<v16bf*>(l) = *reinterpret_cast<const v16bf*>(g);
#endif
  };

  stage(0, 0);
  v8f acc[4] = {{}, {}, {}, {}};
  for (int kt = 0; kt < KT; ++kt) {
    int buf = kt & 1;
#if MOG_ASYNC_LDS
    __builtin_amdgcn_s_wait_asynccnt(0);   // own async writes into btile[buf] done
#endif
    __syncthreads();                       // all waves' writes visible; prior reads of buf^1 done
    if (kt + 1 < KT) stage(kt + 1, buf ^ 1);
    v16bf a = aP[(long)kt * 32];
#pragma unroll
    for (int i = 0; i < 4; ++i) {
      v16bf b = *reinterpret_cast<const v16bf*>(&btile[buf][wave_n * 4 + i][(long)lane * 16]);
      acc[i] = __builtin_amdgcn_wmma_f32_16x16x32_bf16(false, a, false, b, (short)0, acc[i], false, false);
    }
  }
  int mloc = (lane & 16) ? 8 : 0;
  int nn = lane & 15;
#pragma unroll
  for (int i = 0; i < 4; ++i) {
    int nt = ntB + wave_n * 4 + i;
#pragma unroll
    for (int r = 0; r < 8; ++r) {
      int mrow = mt * 16 + r + mloc;        // row m = s*8 + b
      int s = mrow >> 3, bb = mrow & 7;
      out[(long)bb * Ssz * Vsz + (long)s * Vsz + (long)nt * 16 + nn] = acc[i][r];
    }
  }
}

// =====================================================================
extern "C" void kernel_launch(void* const* d_in, const int* in_sizes, int n_in,
                              void* d_out, int out_size, void* d_ws, size_t ws_size,
                              hipStream_t stream) {
  (void)n_in; (void)out_size; (void)ws_size;
  struct LayerP {
    const float *mogW[Rsz], *mogb[Rsz];
    const float *WW, *Wb, *UW, *Ub, *b, *gamma, *beta;
    const float *f1W, *f1b, *f2W, *f2b, *oW, *ob;
  } Ls[Lsz];
  const int* xids; const float* embed;

  if (in_sizes[0] == Bsz * Ssz) {
    // insertion-order flatten: x, embed, then each layer in declaration order
    int i = 0;
    xids  = (const int*)d_in[i++];
    embed = (const float*)d_in[i++];
    for (int m = 0; m < Lsz; ++m) {
      for (int r = 0; r < Rsz; ++r) { Ls[m].mogW[r] = (const float*)d_in[i++]; Ls[m].mogb[r] = (const float*)d_in[i++]; }
      Ls[m].WW = (const float*)d_in[i++]; Ls[m].Wb = (const float*)d_in[i++];
      Ls[m].UW = (const float*)d_in[i++]; Ls[m].Ub = (const float*)d_in[i++];
      Ls[m].b = (const float*)d_in[i++]; Ls[m].gamma = (const float*)d_in[i++]; Ls[m].beta = (const float*)d_in[i++];
      Ls[m].f1W = (const float*)d_in[i++]; Ls[m].f1b = (const float*)d_in[i++];
      Ls[m].f2W = (const float*)d_in[i++]; Ls[m].f2b = (const float*)d_in[i++];
      Ls[m].oW  = (const float*)d_in[i++]; Ls[m].ob  = (const float*)d_in[i++];
    }
  } else {
    // jax tree (alphabetical dict keys): params{embed, layers[ U,W,b,beta,ffw1,ffw2,gamma,mog,o_proj ]}, x
    int i = 0;
    embed = (const float*)d_in[i++];
    for (int m = 0; m < Lsz; ++m) {
      Ls[m].UW = (const float*)d_in[i++]; Ls[m].Ub = (const float*)d_in[i++];
      Ls[m].WW = (const float*)d_in[i++]; Ls[m].Wb = (const float*)d_in[i++];
      Ls[m].b = (const float*)d_in[i++]; Ls[m].beta = (const float*)d_in[i++];
      Ls[m].f1W = (const float*)d_in[i++]; Ls[m].f1b = (const float*)d_in[i++];
      Ls[m].f2W = (const float*)d_in[i++]; Ls[m].f2b = (const float*)d_in[i++];
      Ls[m].gamma = (const float*)d_in[i++];
      for (int r = 0; r < Rsz; ++r) { Ls[m].mogW[r] = (const float*)d_in[i++]; Ls[m].mogb[r] = (const float*)d_in[i++]; }
      Ls[m].oW = (const float*)d_in[i++]; Ls[m].ob = (const float*)d_in[i++];
    }
    xids = (const int*)d_in[i++];
  }

  // ---------- workspace bump allocator ----------
  char* wp = (char*)d_ws;
  auto alloc = [&](size_t bytes) -> char* {
    char* r = wp; wp += (bytes + 255) & ~(size_t)255; return r;
  };
  __bf16 *PM[Lsz][Rsz], *PW[Lsz], *PU[Lsz], *PF1[Lsz], *PF2[Lsz], *PO[Lsz];
  for (int m = 0; m < Lsz; ++m) {
    for (int r = 0; r < Rsz; ++r) PM[m][r] = (__bf16*)alloc((size_t)Hsz * Hsz * 2);
    PW[m]  = (__bf16*)alloc((size_t)Hsz * 4 * Hsz * 2);
    PU[m]  = (__bf16*)alloc((size_t)Hsz * 4 * Hsz * 2);
    PF1[m] = (__bf16*)alloc((size_t)Hsz * 4 * Hsz * 2);
    PF2[m] = (__bf16*)alloc((size_t)4 * Hsz * Hsz * 2);
    PO[m]  = (__bf16*)alloc((size_t)Hsz * Hsz * 2);
  }
  __bf16* PE   = (__bf16*)alloc((size_t)Vsz * Hsz * 2);
  float*  Xemb = (float*)alloc((size_t)Ssz * Bsz * Hsz * 4);
  float*  ys   = (float*)alloc((size_t)Ssz * Bsz * Hsz * 4);
  __bf16* ysP  = (__bf16*)alloc((size_t)Ssz * Bsz * Hsz * 2);
  float*  cst  = (float*)alloc((size_t)2 * Lsz * Bsz * Hsz * 4);  // c then h
  float*  hstA = cst + (size_t)Lsz * Bsz * Hsz;
  float*  xnorm = (float*)alloc((size_t)Bsz * Hsz * 4);
  float*  xm    = (float*)alloc((size_t)Bsz * Hsz * 4);
  float*  hm    = (float*)alloc((size_t)Bsz * Hsz * 4);
  float*  lin   = (float*)alloc((size_t)Bsz * 4 * Hsz * 4);
  float*  ffh   = (float*)alloc((size_t)Bsz * 4 * Hsz * 4);
  float*  res0  = (float*)alloc((size_t)Bsz * Hsz * 4);

  // Packed-A scratch (bf16 fragment layout, 16 rows padded; rows 8..15 zeroed once).
  const size_t PA_1K = 16384, PA_4K = 65536;
  size_t pkHalfs = PA_1K * 3 + PA_4K + PA_1K * Lsz;
  __bf16* pkBase  = (__bf16*)alloc(pkHalfs * 2);
  __bf16* xnormP  = pkBase;
  __bf16* xmP     = xnormP + PA_1K;
  __bf16* hmP     = xmP + PA_1K;
  __bf16* ffhP    = hmP + PA_1K;
  __bf16* hSP[Lsz];
  hSP[0] = ffhP + PA_4K;
  hSP[1] = hSP[0] + PA_1K;

  auto gblk = [](long total) { long b = (total + 255) / 256; return (int)(b > 32768 ? 32768 : b); };

  // ---------- one-time: zero state + packed scratch, gather embeddings, pack weights ----------
  mog_zero<<<gblk((long)2 * Lsz * Bsz * Hsz), 256, 0, stream>>>(cst, (long)2 * Lsz * Bsz * Hsz);
  mog_zero<<<gblk((long)(pkHalfs / 2)), 256, 0, stream>>>((float*)pkBase, (long)(pkHalfs / 2));
  mog_gather<<<gblk((long)Ssz * Bsz * Hsz), 256, 0, stream>>>(xids, embed, Xemb, (long)Ssz * Bsz * Hsz);
  for (int m = 0; m < Lsz; ++m) {
    for (int r = 0; r < Rsz; ++r)
      mog_pack_B<<<gblk((long)Hsz * Hsz), 256, 0, stream>>>(Ls[m].mogW[r], PM[m][r], (long)Hsz * Hsz, Hsz, Hsz, 0);
    mog_pack_B<<<gblk((long)Hsz * 4 * Hsz), 256, 0, stream>>>(Ls[m].WW, PW[m], (long)Hsz * 4 * Hsz, Hsz, 4 * Hsz, 0);
    mog_pack_B<<<gblk((long)Hsz * 4 * Hsz), 256, 0, stream>>>(Ls[m].UW, PU[m], (long)Hsz * 4 * Hsz, Hsz, 4 * Hsz, 0);
    mog_pack_B<<<gblk((long)Hsz * 4 * Hsz), 256, 0, stream>>>(Ls[m].f1W, PF1[m], (long)Hsz * 4 * Hsz, Hsz, 4 * Hsz, 0);
    mog_pack_B<<<gblk((long)4 * Hsz * Hsz), 256, 0, stream>>>(Ls[m].f2W, PF2[m], (long)4 * Hsz * Hsz, 4 * Hsz, Hsz, 0);
    mog_pack_B<<<gblk((long)Hsz * Hsz), 256, 0, stream>>>(Ls[m].oW, PO[m], (long)Hsz * Hsz, Hsz, Hsz, 0);
  }
  mog_pack_B<<<gblk((long)Vsz * Hsz), 256, 0, stream>>>(embed, PE, (long)Vsz * Hsz, Hsz, Vsz, 1);  // B = embed^T

  auto gemmBlocks = [](int N) { return (N / 16 * 32 + 255) / 256; };

  // ---------- sequential scan ----------
  for (int t = 0; t < Ssz; ++t) {
    const float* x_in = Xemb + (size_t)t * Bsz * Hsz;
    for (int m = 0; m < Lsz; ++m) {
      const float* inp  = (m == 0) ? x_in : res0;
      const float* prev = x_in;  // lagged residual: prev_input for layer0 and layer1 is x_in (L=2)
      float* cm = cst  + (size_t)m * Bsz * Hsz;
      float* hS = hstA + (size_t)m * Bsz * Hsz;

      mog_ln8<<<1, 256, 0, stream>>>(inp, Ls[m].gamma, Ls[m].beta, xnorm, xnormP);

      const __bf16* Aop[Rsz] = { xnormP, hmP, xmP, hmP, xmP };
      const float*  mul[Rsz] = { hS, xnorm, hm, xm, hm };
      float*        dst[Rsz] = { hm, xm, hm, xm, hm };
      __bf16*       dsP[Rsz] = { hmP, xmP, hmP, xmP, hmP };
      for (int r = 0; r < Rsz; ++r)
        mog_smallgemm<<<gemmBlocks(Hsz), 256, 0, stream>>>(
            Aop[r], PM[m][r], Hsz, nullptr, nullptr, 0,
            Ls[m].mogb[r], nullptr, nullptr, nullptr, nullptr, mul[r],
            dst[r], dsP[r], Hsz, 2);

      mog_smallgemm<<<gemmBlocks(4 * Hsz), 256, 0, stream>>>(
          xmP, PW[m], Hsz, hmP, PU[m], Hsz,
          Ls[m].Wb, Ls[m].Ub, Ls[m].b, nullptr, nullptr, nullptr,
          lin, nullptr, 4 * Hsz, 0);

      mog_cell<<<(Bsz * Hsz) / 256, 256, 0, stream>>>(lin, cm, hS, hSP[m]);

      mog_smallgemm<<<gemmBlocks(4 * Hsz), 256, 0, stream>>>(
          xnormP, PF1[m], Hsz, nullptr, nullptr, 0,
          Ls[m].f1b, nullptr, nullptr, nullptr, nullptr, nullptr,
          ffh, ffhP, 4 * Hsz, 1);

      float* outp = (m == Lsz - 1) ? (ys + (size_t)t * Bsz * Hsz) : res0;
      mog_smallgemm<<<gemmBlocks(Hsz), 256, 0, stream>>>(
          ffhP, PF2[m], 4 * Hsz, hSP[m], PO[m], Hsz,
          Ls[m].f2b, Ls[m].ob, nullptr, xm, prev, nullptr,
          outp, nullptr, Hsz, 3);
    }
  }

  // ---------- logits: pack ys as A operand, then big WMMA GEMM ----------
  mog_pack_A<<<gblk((long)Ssz * Bsz * Hsz), 256, 0, stream>>>(ys, ysP, (long)Ssz * Bsz * Hsz);
  dim3 lgrid(Vsz / 128, (Ssz * Bsz) / 64);  // 250 x 32
  mog_logits<<<lgrid, 256, 0, stream>>>(ysP, PE, (float*)d_out);
}
